// PointMambaBlock_55413668053380
// MI455X (gfx1250) — matrix operand order repaired and checked by reading, not compiled, not run on hardware
//
#include <hip/hip_runtime.h>
#include <hip/hip_bf16.h>

// ---------------- problem constants (from reference) ----------------
constexpr int NPTS  = 32768;
constexpr int CCH   = 192;   // C
constexpr int DI    = 384;
constexpr int DSZ   = 16;    // DS (state dim)
constexpr int DRNK  = 12;    // DR (dt rank)
constexpr int KC    = 4;     // conv kernel
constexpr int KN    = 27;    // neighbors
constexpr int PATCH = 64;
constexpr int LSEQ  = 512;   // sequence length for both branches
constexpr int EDIM  = DRNK + 2 * DSZ;  // 44
constexpr int EPAD  = 48;              // padded xdbl row stride

typedef __attribute__((ext_vector_type(16))) __bf16 v16bf;
typedef __attribute__((ext_vector_type(8)))  float  v8f;

// -------- helpers --------
// packed f32x2 -> bf16x2 (round-to-nearest-even) in one VALU op
static __device__ __forceinline__ unsigned pk_bf16(float lo, float hi) {
  unsigned r;
  asm("v_cvt_pk_bf16_f32 %0, %1, %2" : "=v"(r) : "v"(lo), "v"(hi));
  return r;
}
union FragBF {
  v16bf v;
  uint4 q[2];
};
static __device__ __forceinline__ float siluf(float x) {
  return x / (1.0f + __expf(-x));
}
static __device__ __forceinline__ float softplusf(float x) {
  if (x > 20.0f) return x;
  if (x < -20.0f) return __expf(x);
  return __logf(1.0f + __expf(x));
}

// ================= 0) f32 -> bf16 conversion pre-passes ======================
// elementwise pair conversion: n2 = number of float2 pairs
__global__ void k_cvt(const float* __restrict__ in, __bf16* __restrict__ out, int n2) {
  int i = blockIdx.x * blockDim.x + threadIdx.x;
  if (i < n2) {
    float2 p = ((const float2*)in)[i];
    ((unsigned*)out)[i] = pk_bf16(p.x, p.y);
  }
}
// weight transpose + convert: W[N,K] f32 -> Wt[K,N] bf16
__global__ void k_cvt_wt(const float* __restrict__ W, __bf16* __restrict__ Wt,
                         int Ncols, int K) {
  int i = blockIdx.x * blockDim.x + threadIdx.x;   // over K * (N/2)
  int nh = Ncols >> 1;
  if (i < K * nh) {
    int k = i / nh, np = i - k * nh;
    int n0 = 2 * np;
    ((unsigned*)Wt)[(size_t)k * nh + np] =
        pk_bf16(W[(size_t)n0 * K + k], W[(size_t)(n0 + 1) * K + k]);
  }
}

// ================= 1) CPE gather + BN partial sums (deterministic) ===========
constexpr int CPE_BLOCKS = 256;
__global__ void k_cpe(const float* __restrict__ data, const int* __restrict__ neigh,
                      const float* __restrict__ cpe_w,
                      float* __restrict__ out, float* __restrict__ psum,
                      float* __restrict__ psumsq) {
  const int c = threadIdx.x;           // 0..191
  float ls = 0.f, ls2 = 0.f;
  for (int n = blockIdx.x; n < NPTS; n += CPE_BLOCKS) {
    float o = 0.f;
    #pragma unroll 1
    for (int k = 0; k < KN; ++k) {
      int idx = neigh[n * KN + k];
      o += data[(size_t)idx * CCH + c] * cpe_w[k * CCH + c];
    }
    out[(size_t)n * CCH + c] = o;
    ls += o; ls2 += o * o;
  }
  psum[blockIdx.x * CCH + c] = ls;
  psumsq[blockIdx.x * CCH + c] = ls2;
}

__global__ void k_bnstats(const float* __restrict__ psum, const float* __restrict__ psumsq,
                          const float* __restrict__ bn_g, const float* __restrict__ bn_b,
                          float* __restrict__ scale, float* __restrict__ shift) {
  const int c = threadIdx.x;
  float s = 0.f, q = 0.f;
  for (int b = 0; b < CPE_BLOCKS; ++b) { s += psum[b * CCH + c]; q += psumsq[b * CCH + c]; }
  float mean = s / (float)NPTS;
  float var  = q / (float)NPTS - mean * mean;
  float sc = bn_g[c] * rsqrtf(var + 1e-5f);
  scale[c] = sc;
  shift[c] = bn_b[c] - mean * sc;
}

__global__ void k_bnres(const float* __restrict__ cpe_out, const float* __restrict__ data,
                        const float* __restrict__ scale, const float* __restrict__ shift,
                        float* __restrict__ d) {
  const int c = threadIdx.x;
  const size_t i = (size_t)blockIdx.x * CCH + c;
  d[i] = cpe_out[i] * scale[c] + shift[c] + data[i];
}

// ================= 2) Row LayerNorm over C=192 (block=256) ===================
__global__ void k_rowln(const float* __restrict__ in, const float* __restrict__ g,
                        const float* __restrict__ b, float* __restrict__ out) {
  __shared__ float s1[256], s2[256];
  const int t = threadIdx.x;
  const size_t row = blockIdx.x;
  float v = (t < CCH) ? in[row * CCH + t] : 0.f;
  s1[t] = v; s2[t] = v * v;
  __syncthreads();
  for (int s = 128; s > 0; s >>= 1) {
    if (t < s) { s1[t] += s1[t + s]; s2[t] += s2[t + s]; }
    __syncthreads();
  }
  float mean = s1[0] / (float)CCH;
  float var  = s2[0] / (float)CCH - mean * mean;
  float rs = rsqrtf(var + 1e-5f);
  if (t < CCH) out[row * CCH + t] = (v - mean) * rs * g[t] + b[t];
}

// ================= 3) bf16 WMMA GEMM: Out[M,N] = A[M,K] * Wt[K,N] ============
// A: bf16 row-major [M,K]; Wt: bf16 [K,N] (pre-transposed weights).
// One wave per block computing a 16M x 64N strip: 4 accumulators share one A
// fragment per K-step. A loads are contiguous 16B vectors; B loads are single bf16 elements strided by Ncols.
// grid = (M/16, Ncols/64), block = 32.
template <bool HAS_BIAS, bool HAS_RES>
__global__ void k_gemm(const __bf16* __restrict__ A, const __bf16* __restrict__ Wt,
                       float* __restrict__ Out, const float* __restrict__ bias,
                       const float* __restrict__ res, int M, int Ncols, int K) {
  const int lane = threadIdx.x & 31;
  const int tm = blockIdx.x;
  const int tn0 = blockIdx.y * 4;
  const int g  = lane >> 4;        // lane group 0/1
  const int ml = lane & 15;
  const int mrow = tm * 16 + ml;
  const __bf16* arow = A + (size_t)mrow * K + g * 8;
  v8f acc0 = {}, acc1 = {}, acc2 = {}, acc3 = {};
  for (int k0 = 0; k0 < K; k0 += 32) {
    FragBF af;
    af.q[0] = *(const uint4*)(arow + k0);        // K = k0+g*8   .. +7
    af.q[1] = *(const uint4*)(arow + k0 + 16);   // K = k0+16+g*8 .. +7
    const __bf16* wrow = Wt + (size_t)(k0 + g * 8) * Ncols + tn0 * 16 + ml;  // B: lane holds column ml, K same order as A
    #pragma unroll
    for (int t = 0; t < 4; ++t) {
      FragBF bf;
      #pragma unroll
      for (int s = 0; s < 8; ++s) {
        bf.v[s]     = wrow[(size_t)s * Ncols + t * 16];
        bf.v[s + 8] = wrow[(size_t)(s + 16) * Ncols + t * 16];
      }
      v8f& acc = (t == 0) ? acc0 : (t == 1) ? acc1 : (t == 2) ? acc2 : acc3;
      acc = __builtin_amdgcn_wmma_f32_16x16x32_bf16(false, af.v, false, bf.v,
                                                    (short)0, acc, false, false);
    }
  }
  #pragma unroll
  for (int t = 0; t < 4; ++t) {
    v8f& acc = (t == 0) ? acc0 : (t == 1) ? acc1 : (t == 2) ? acc2 : acc3;
    #pragma unroll
    for (int r = 0; r < 8; ++r) {
      int row = tm * 16 + r + 8 * g;
      int col = (tn0 + t) * 16 + ml;
      float v = acc[r];
      if (HAS_BIAS) v += bias[col];
      if (HAS_RES)  v += res[(size_t)row * Ncols + col];
      Out[(size_t)row * Ncols + col] = v;
    }
  }
}

// ================= 4) depthwise causal conv (K=4) + SiLU =====================
__global__ void k_conv(const float* __restrict__ xz, const float* __restrict__ cw,
                       const float* __restrict__ cb, float* __restrict__ xsi,
                       int stride, int isbwd) {
  const int l = blockIdx.x;
  const int d = threadIdx.x;             // 0..383
  const int j = l / stride;              // sequence index in original order
  float acc = cb[d];
  #pragma unroll
  for (int k = 0; k < KC; ++k) {
    int dj = isbwd ? (3 - k) : (k - 3);
    int jn = j + dj;
    if (jn >= 0 && jn < LSEQ)
      acc += cw[d * KC + k] * xz[(size_t)(l + dj * stride) * (2 * DI) + d];
  }
  xsi[(size_t)l * DI + d] = siluf(acc);
}

// ================= 5) W_x projection: xdbl[l,e] = xsi[l,:] . W_x[e,:] ========
__global__ void k_wx(const float* __restrict__ xsi, const float* __restrict__ Wx,
                     float* __restrict__ xdbl) {
  __shared__ float srow[DI];
  const int l = blockIdx.x;
  const int t = threadIdx.x;             // 64 threads
  for (int i = t; i < DI; i += 64) srow[i] = xsi[(size_t)l * DI + i];
  __syncthreads();
  if (t < EDIM) {
    float acc = 0.f;
    const float* wr = &Wx[t * DI];
    for (int d = 0; d < DI; ++d) acc += srow[d] * wr[d];
    xdbl[(size_t)l * EPAD + t] = acc;
  }
}

// ================= 6) selective scan (one block = one sequence) ==============
__global__ void k_scan(const float* __restrict__ xsi, const float* __restrict__ xdbl,
                       const float* __restrict__ xz, const float* __restrict__ W_dt,
                       const float* __restrict__ b_dt, const float* __restrict__ A_log,
                       const float* __restrict__ Dp, float* __restrict__ y,
                       int stride, int isbwd, int accumulate) {
  __shared__ float sh[EDIM];
  const int d = threadIdx.x;
  const int base = blockIdx.x;           // pixel: p in [0,64); window: 0
  float wdt[DRNK];
  #pragma unroll
  for (int r = 0; r < DRNK; ++r) wdt[r] = W_dt[d * DRNK + r];
  float Arow[DSZ];
  #pragma unroll
  for (int s = 0; s < DSZ; ++s) Arow[s] = -__expf(A_log[d * DSZ + s]);
  const float bdt = b_dt[d];
  const float Dd  = Dp[d];
  float h[DSZ];
  #pragma unroll
  for (int s = 0; s < DSZ; ++s) h[s] = 0.f;

  for (int j = 0; j < LSEQ; ++j) {
    const int jj = isbwd ? (LSEQ - 1 - j) : j;
    const int l = base + jj * stride;
    __syncthreads();
    if (d < EDIM) sh[d] = xdbl[(size_t)l * EPAD + d];
    __syncthreads();
    float dtr = bdt;
    #pragma unroll
    for (int r = 0; r < DRNK; ++r) dtr += wdt[r] * sh[r];
    const float dt = softplusf(dtr);
    const float xv = xsi[(size_t)l * DI + d];
    const float dtx = dt * xv;
    float ys = 0.f;
    #pragma unroll
    for (int s = 0; s < DSZ; ++s) {
      h[s] = __expf(dt * Arow[s]) * h[s] + dtx * sh[DRNK + s];
      ys += h[s] * sh[DRNK + DSZ + s];
    }
    const float zv = xz[(size_t)l * (2 * DI) + DI + d];
    const float o = (ys + Dd * xv) * siluf(zv);
    float* yp = &y[(size_t)l * DI + d];
    *yp = accumulate ? (*yp + o) : o;
  }
}

// ================= 7) pixel epilogue permute + residual ======================
__global__ void k_permadd(const float* __restrict__ tokout, const float* __restrict__ xln1,
                          float* __restrict__ out_pix) {
  const int lp = blockIdx.x;
  const int c = threadIdx.x;
  const int p = lp / LSEQ, j = lp % LSEQ;
  const int ltok = j * PATCH + p;
  out_pix[(size_t)lp * CCH + c] = tokout[(size_t)ltok * CCH + c] + xln1[(size_t)lp * CCH + c];
}

// ================= 8) window patch mean ======================================
__global__ void k_patchmean(const float* __restrict__ pix, float* __restrict__ xd) {
  const int j = blockIdx.x;   // 0..511
  const int c = threadIdx.x;
  float s = 0.f;
  for (int p = 0; p < PATCH; ++p) s += pix[(size_t)(j * PATCH + p) * CCH + c];
  xd[(size_t)j * CCH + c] = s * (1.0f / PATCH);
}

// ================= 9) window repeat + residual ===============================
__global__ void k_repeatadd(const float* __restrict__ xm, const float* __restrict__ pix,
                            float* __restrict__ out) {
  const int l = blockIdx.x;
  const int c = threadIdx.x;
  out[(size_t)l * CCH + c] = xm[(size_t)(l >> 6) * CCH + c] + pix[(size_t)l * CCH + c];
}

// =============================================================================
struct MP {
  const float *W_in, *W_out, *ln_g, *ln_b;
  const float *conv_w[2], *conv_b[2], *W_x[2], *W_dt[2], *b_dt[2], *A_log[2], *D[2];
};
static void fill_mp(MP& m, void* const* d_in, int base) {
  m.W_in  = (const float*)d_in[base + 0];
  m.W_out = (const float*)d_in[base + 1];
  m.ln_g  = (const float*)d_in[base + 2];
  m.ln_b  = (const float*)d_in[base + 3];
  for (int s = 0; s < 2; ++s) {
    int b = base + 4 + 7 * s;
    m.conv_w[s] = (const float*)d_in[b + 0];
    m.conv_b[s] = (const float*)d_in[b + 1];
    m.W_x[s]    = (const float*)d_in[b + 2];
    m.W_dt[s]   = (const float*)d_in[b + 3];
    m.b_dt[s]   = (const float*)d_in[b + 4];
    m.A_log[s]  = (const float*)d_in[b + 5];
    m.D[s]      = (const float*)d_in[b + 6];
  }
}

// host-side helper: convert A (f32, M*K) and W (f32, NxK) then run WMMA GEMM
template <bool HAS_BIAS, bool HAS_RES>
static void run_gemm(const float* A, const float* W, float* Out,
                     const float* bias, const float* res, int M, int Ncols, int K,
                     __bf16* aBF, __bf16* wBF, hipStream_t stream) {
  int n2a = (M * K) >> 1;
  k_cvt<<<(n2a + 255) / 256, 256, 0, stream>>>(A, aBF, n2a);
  int nw = K * (Ncols >> 1);
  k_cvt_wt<<<(nw + 255) / 256, 256, 0, stream>>>(W, wBF, Ncols, K);
  k_gemm<HAS_BIAS, HAS_RES><<<dim3(M / 16, Ncols / 64), 32, 0, stream>>>(
      aBF, wBF, Out, bias, res, M, Ncols, K);
}

extern "C" void kernel_launch(void* const* d_in, const int* in_sizes, int n_in,
                              void* d_out, int out_size, void* d_ws, size_t ws_size,
                              hipStream_t stream) {
  (void)in_sizes; (void)n_in; (void)out_size; (void)ws_size;
  const float* data  = (const float*)d_in[0];
  const int*   neigh = (const int*)d_in[1];
  const float* cpe_w = (const float*)d_in[3];
  const float* bn_g  = (const float*)d_in[4];
  const float* bn_b  = (const float*)d_in[5];
  const float* ln1_g = (const float*)d_in[6];
  const float* ln1_b = (const float*)d_in[7];
  MP px, wn;
  fill_mp(px, d_in, 8);
  fill_mp(wn, d_in, 26);
  const float* proj_w = (const float*)d_in[44];
  const float* proj_b = (const float*)d_in[45];
  float* out = (float*)d_out;

  float* ws = (float*)d_ws;
  const size_t NC = (size_t)NPTS * CCH;
  size_t oD  = 0;                         // d residual
  size_t oXL = oD  + NC;                  // ln1 output
  size_t oXZ = oXL + NC;                  // xz N x 768
  size_t oXS = oXZ + (size_t)NPTS * (2 * DI); // xsi N x 384 (later: window output)
  size_t oY  = oXS + (size_t)NPTS * DI;   // y accumulate N x 384
  size_t oDB = oY  + (size_t)NPTS * DI;   // xdbl N x 48
  size_t oT6 = oDB + (size_t)NPTS * EPAD; // multi: cpe_out -> tokout -> window bufs
  size_t oT7 = oT6 + NC;                  // multi: xn2 -> out_pix
  size_t oST = oT7 + NC;                  // stats
  size_t oBF = oST + (size_t)2 * CPE_BLOCKS * CCH + 1024;  // bf16 scratch (float units)
  float* psum   = ws + oST;
  float* psumsq = psum + (size_t)CPE_BLOCKS * CCH;
  float* scale  = psumsq + (size_t)CPE_BLOCKS * CCH;
  float* shift  = scale + CCH;
  __bf16* aBF = (__bf16*)(ws + oBF);                     // N x 384 bf16 max
  __bf16* wBF = (__bf16*)(ws + oBF + (size_t)NPTS * DI / 2); // 192x768 bf16 max
  // window sub-buffers inside oT6 (free after pixel tokout consumed)
  float* w_xd  = ws + oT6;
  float* w_tok = w_xd  + (size_t)LSEQ * CCH;
  float* w_xz  = w_tok + (size_t)LSEQ * CCH;
  float* w_xsi = w_xz  + (size_t)LSEQ * (2 * DI);
  float* w_db  = w_xsi + (size_t)LSEQ * DI;
  float* w_y   = w_db  + (size_t)LSEQ * EPAD;
  float* w_xm  = w_y   + (size_t)LSEQ * DI;

  // ---- CPE + BN + residual ----
  k_cpe<<<CPE_BLOCKS, CCH, 0, stream>>>(data, neigh, cpe_w, ws + oT6, psum, psumsq);
  k_bnstats<<<1, CCH, 0, stream>>>(psum, psumsq, bn_g, bn_b, scale, shift);
  k_bnres<<<NPTS, CCH, 0, stream>>>(ws + oT6, data, scale, shift, ws + oD);
  // ---- LN1 and pixel-branch inner LN ----
  k_rowln<<<NPTS, 256, 0, stream>>>(ws + oD, ln1_g, ln1_b, ws + oXL);
  k_rowln<<<NPTS, 256, 0, stream>>>(ws + oXL, px.ln_g, px.ln_b, ws + oT7);
  // ---- pixel branch ----
  run_gemm<false, false>(ws + oT7, px.W_in, ws + oXZ, nullptr, nullptr,
                         NPTS, 2 * DI, CCH, aBF, wBF, stream);
  for (int dir = 0; dir < 2; ++dir) {
    k_conv<<<NPTS, DI, 0, stream>>>(ws + oXZ, px.conv_w[dir], px.conv_b[dir],
                                    ws + oXS, PATCH, dir);
    k_wx<<<NPTS, 64, 0, stream>>>(ws + oXS, px.W_x[dir], ws + oDB);
    k_scan<<<PATCH, DI, 0, stream>>>(ws + oXS, ws + oDB, ws + oXZ, px.W_dt[dir],
                                     px.b_dt[dir], px.A_log[dir], px.D[dir],
                                     ws + oY, PATCH, dir, dir);
  }
  run_gemm<false, false>(ws + oY, px.W_out, ws + oT6, nullptr, nullptr,
                         NPTS, CCH, DI, aBF, wBF, stream);
  k_permadd<<<NPTS, CCH, 0, stream>>>(ws + oT6, ws + oXL, ws + oT7);
  // ---- window branch ----
  k_patchmean<<<LSEQ, CCH, 0, stream>>>(ws + oT7, w_xd);
  k_rowln<<<LSEQ, 256, 0, stream>>>(w_xd, wn.ln_g, wn.ln_b, w_tok);
  run_gemm<false, false>(w_tok, wn.W_in, w_xz, nullptr, nullptr,
                         LSEQ, 2 * DI, CCH, aBF, wBF, stream);
  for (int dir = 0; dir < 2; ++dir) {
    k_conv<<<LSEQ, DI, 0, stream>>>(w_xz, wn.conv_w[dir], wn.conv_b[dir],
                                    w_xsi, 1, dir);
    k_wx<<<LSEQ, 64, 0, stream>>>(w_xsi, wn.W_x[dir], w_db);
    k_scan<<<1, DI, 0, stream>>>(w_xsi, w_db, w_xz, wn.W_dt[dir], wn.b_dt[dir],
                                 wn.A_log[dir], wn.D[dir], w_y, 1, dir, dir);
  }
  run_gemm<false, false>(w_y, wn.W_out, w_xm, nullptr, nullptr,
                         LSEQ, CCH, DI, aBF, wBF, stream);
  k_repeatadd<<<NPTS, CCH, 0, stream>>>(w_xm, ws + oT7, ws + oXS);
  // ---- final projection with fused bias + residual -> d_out ----
  run_gemm<true, true>(ws + oXS, proj_w, out, proj_b, ws + oD,
                       NPTS, CCH, CCH, aBF, wBF, stream);
}
